// GraphUNet_111669150312
// MI455X (gfx1250) — matrix operand order repaired and checked by reading, not compiled, not run on hardware
//
#include <hip/hip_runtime.h>

// ---------------------------------------------------------------------------
// GraphUNet (N=4096, F=64, DEPTH=3, ratio=0.5) on MI455X (gfx1250, wave32).
// Dominant work = dense A_sl@A_sl augment GEMMs -> v_wmma_f32_16x16x32_f16.
// Adjacency kept in f16 (exact 0/1 / small-count values), f32 accumulate.
// B tiles staged TRANSPOSED in LDS so both A and B WMMA fragments are
// contiguous ds_load_b128 pairs.  A-tile staging uses the CDNA5 async
// global->LDS path (ASYNCcnt tracked) via __builtin_amdgcn_global_load_
// async_to_lds_b128.
// ---------------------------------------------------------------------------

typedef __attribute__((ext_vector_type(16))) _Float16 v16h;
typedef __attribute__((ext_vector_type(8)))  float    v8f;
typedef int b128_t __attribute__((vector_size(16)));   // matches builtin param

#define WMMA_F16(a, b, c) \
  __builtin_amdgcn_wmma_f32_16x16x32_f16(false, (a), false, (b), (short)0, (c), false, false)

#if defined(__gfx1250__) && __has_builtin(__builtin_amdgcn_global_load_async_to_lds_b128)
#define HAVE_ASYNC_LDS 1
#else
#define HAVE_ASYNC_LDS 0
#endif

#define GLOBAL_AS __attribute__((address_space(1)))
#define LDS_AS    __attribute__((address_space(3)))

// 16-byte global -> LDS copy (async if available; falls back to load+store).
__device__ __forceinline__ void async_copy16(const _Float16* g, _Float16* l) {
#if HAVE_ASYNC_LDS
  __builtin_amdgcn_global_load_async_to_lds_b128(
      (GLOBAL_AS b128_t*)g, (LDS_AS b128_t*)l, 0, 0);
#else
  *(uint4*)l = *(const uint4*)g;
#endif
}

__device__ __forceinline__ void wait_async() {
#if HAVE_ASYNC_LDS
#if __has_builtin(__builtin_amdgcn_s_wait_asynccnt)
  __builtin_amdgcn_s_wait_asynccnt(0);
#else
  asm volatile("s_wait_asynccnt 0x0" ::: "memory");
#endif
#endif
}

// ----- WMMA fragment loaders (contiguous k = 16*hi + s order, wave32) -----------
// A (16x32 f16) from row-major LDS: lanes 0-15 = M rows; VGPRs 0-7 hold
// K=hi*16+0..15 (hi = lane>=16), matching frag_bt_lds.
__device__ __forceinline__ v16h frag_a_lds(const _Float16* sA, int lda, int mbase, int lane) {
  const _Float16* p = sA + (mbase + (lane & 15)) * lda + (lane >> 4) * 16;
  v16h f;
#pragma unroll
  for (int j = 0; j < 8; ++j) f[j] = p[j];
#pragma unroll
  for (int j = 0; j < 8; ++j) f[8 + j] = p[8 + j];
  return f;
}

// B (32x16 f16) from COLUMN-MAJOR LDS (sBt[col*ldt + k]): lane wants
// f[j] = B[koff + j][nc]  ->  16 contiguous halfs = 2x ds_load_b128.
__device__ __forceinline__ v16h frag_bt_lds(const _Float16* sBt, int ldt, int nbase, int lane) {
  const _Float16* p = sBt + (nbase + (lane & 15)) * ldt + (lane >> 4) * 16;
  v16h f;
#pragma unroll
  for (int j = 0; j < 16; ++j) f[j] = p[j];
  return f;
}

union H8 { uint4 q; _Float16 h[8]; };

// ---------------------------------------------------------------------------
// Augment GEMM:  C_f16 = (A + I)(A + I) with zero diagonal.  A: n x n f16.
// Block tile 128x128, K-step 32, 256 threads = 8 waves (2x4 wave grid).
// ---------------------------------------------------------------------------
#define AG_LD 40

__global__ __launch_bounds__(256)
void k_augment_gemm(const _Float16* __restrict__ A, _Float16* __restrict__ C, int n) {
  __shared__ _Float16 sA [128 * AG_LD];   // A tile, row-major
  __shared__ _Float16 sBt[128 * AG_LD];   // B tile, column-major (transposed)

  const int tid  = threadIdx.x;
  const int lane = tid & 31;
  const int wid  = tid >> 5;
  const int wm   = wid >> 2;       // 0..1
  const int wn   = wid & 3;        // 0..3
  const int bm0  = blockIdx.y * 128;
  const int bn0  = blockIdx.x * 128;

  v8f acc[4][2] = {};

  const int arow = tid >> 1, acol = (tid & 1) * 16;  // A tile 128x32 copy
  const int bk2  = (tid & 15) * 2;                   // B tile: row pair k2,k2+1
  const int bc8  = (tid >> 4) * 8;                   // B tile: 8-col group

  for (int k0 = 0; k0 < n; k0 += 32) {
    __syncthreads();
    { // stage A tile (rows bm0.., cols k0..): straight copy -> async path
      const _Float16* g = A + (size_t)(bm0 + arow) * n + k0 + acol;
      _Float16* l = sA + arow * AG_LD + acol;
      async_copy16(g, l);
      async_copy16(g + 8, l + 8);
      if (k0 + 32 < n) __builtin_prefetch(g + 32, 0, 1);
    }
    { // stage B tile (rows k0.., cols bn0..) transposed into sBt
      const _Float16* g0 = A + (size_t)(k0 + bk2) * n + bn0 + bc8;
      H8 r0, r1;
      r0.q = *(const uint4*)g0;
      r1.q = *(const uint4*)(g0 + n);
#pragma unroll
      for (int i = 0; i < 8; ++i) {
        union { _Float16 h[2]; unsigned u; } w;
        w.h[0] = r0.h[i];
        w.h[1] = r1.h[i];
        *(unsigned*)(sBt + (bc8 + i) * AG_LD + bk2) = w.u;
      }
      if (k0 + 32 < n) __builtin_prefetch(g0 + (size_t)32 * n, 0, 1);
    }
    wait_async();
    __syncthreads();
    if (tid < 32) {  // A_sl = A + I : patch diagonal in both staged tiles
      const int gk = k0 + tid;
      const int ra = gk - bm0;
      if (ra >= 0 && ra < 128) sA[ra * AG_LD + tid] += (_Float16)1.0f;
      const int cb = gk - bn0;
      if (cb >= 0 && cb < 128) sBt[cb * AG_LD + tid] += (_Float16)1.0f;
    }
    __syncthreads();

    v16h af[4], bf[2];
#pragma unroll
    for (int mt = 0; mt < 4; ++mt) af[mt] = frag_a_lds(sA, AG_LD, wm * 64 + mt * 16, lane);
#pragma unroll
    for (int nt = 0; nt < 2; ++nt) bf[nt] = frag_bt_lds(sBt, AG_LD, wn * 32 + nt * 16, lane);
#pragma unroll
    for (int mt = 0; mt < 4; ++mt)
#pragma unroll
      for (int nt = 0; nt < 2; ++nt)
        acc[mt][nt] = WMMA_F16(af[mt], bf[nt], acc[mt][nt]);
  }

  // Epilogue: f16 store, remove self loops (zero diagonal).
  const int hi = lane >> 4, nlo = lane & 15;
#pragma unroll
  for (int mt = 0; mt < 4; ++mt)
#pragma unroll
    for (int nt = 0; nt < 2; ++nt) {
      const int gn = bn0 + wn * 32 + nt * 16 + nlo;
#pragma unroll
      for (int r = 0; r < 8; ++r) {
        const int gm = bm0 + wm * 64 + mt * 16 + 8 * hi + r;
        C[(size_t)gm * n + gn] = (gm == gn) ? (_Float16)0.0f : (_Float16)acc[mt][nt][r];
      }
    }
}

// ---------------------------------------------------------------------------
// Skinny GEMM: C_f32(M x 64) = A(M x K) @ B(K x 64).  Block tile 64 rows,
// 128 threads = 4 waves; wave w owns rows w*16..w*16+15 across all 4 N-tiles.
// Operands optionally f32 in global (stage-converted to f16 in LDS).
// ---------------------------------------------------------------------------
#define SK_LD 40

template <bool A_F32, bool B_F32>
__global__ __launch_bounds__(128)
void k_gemm_n64(const void* __restrict__ Aop, const void* __restrict__ Bop,
                float* __restrict__ C, int M, int K, int lda) {
  __shared__ _Float16 sA [64 * SK_LD];    // A tile, row-major
  __shared__ _Float16 sBt[64 * SK_LD];    // B tile, column-major

  const int tid  = threadIdx.x;
  const int lane = tid & 31;
  const int wid  = tid >> 5;       // 0..3 -> m-tile
  const int bm0  = blockIdx.x * 64;

  v8f acc[4] = {};

  const int arow = tid >> 1, acol = (tid & 1) * 16;  // A tile 64x32
  const int bk2  = (tid & 15) * 2;                   // B tile: row pair
  const int bc8  = (tid >> 4) * 8;                   // B tile: 8-col group

  for (int k0 = 0; k0 < K; k0 += 32) {
    __syncthreads();
    if (A_F32) {
      const float* g = (const float*)Aop + (size_t)(bm0 + arow) * lda + k0 + acol;
      _Float16* d = sA + arow * SK_LD + acol;
#pragma unroll
      for (int j = 0; j < 16; ++j) d[j] = (_Float16)g[j];
    } else {
      const _Float16* g = (const _Float16*)Aop + (size_t)(bm0 + arow) * lda + k0 + acol;
      _Float16* l = sA + arow * SK_LD + acol;
      async_copy16(g, l);
      async_copy16(g + 8, l + 8);
      if (k0 + 32 < K) __builtin_prefetch(g + 32, 0, 1);
    }
    if (B_F32) {
      const float* g0 = (const float*)Bop + (size_t)(k0 + bk2) * 64 + bc8;
      const float* g1 = g0 + 64;
#pragma unroll
      for (int i = 0; i < 8; ++i) {
        union { _Float16 h[2]; unsigned u; } w;
        w.h[0] = (_Float16)g0[i];
        w.h[1] = (_Float16)g1[i];
        *(unsigned*)(sBt + (bc8 + i) * SK_LD + bk2) = w.u;
      }
    } else {
      const _Float16* g0 = (const _Float16*)Bop + (size_t)(k0 + bk2) * 64 + bc8;
      H8 r0, r1;
      r0.q = *(const uint4*)g0;
      r1.q = *(const uint4*)(g0 + 64);
#pragma unroll
      for (int i = 0; i < 8; ++i) {
        union { _Float16 h[2]; unsigned u; } w;
        w.h[0] = r0.h[i];
        w.h[1] = r1.h[i];
        *(unsigned*)(sBt + (bc8 + i) * SK_LD + bk2) = w.u;
      }
    }
    wait_async();
    __syncthreads();

    v16h af = frag_a_lds(sA, SK_LD, wid * 16, lane);
#pragma unroll
    for (int nt = 0; nt < 4; ++nt) {
      v16h bf = frag_bt_lds(sBt, SK_LD, nt * 16, lane);
      acc[nt] = WMMA_F16(af, bf, acc[nt]);
    }
    __syncthreads();
  }

  const int hi = lane >> 4, nlo = lane & 15;
#pragma unroll
  for (int nt = 0; nt < 4; ++nt)
#pragma unroll
    for (int r = 0; r < 8; ++r) {
      const int gm = bm0 + wid * 16 + 8 * hi + r;
      C[(size_t)gm * 64 + nt * 16 + nlo] = acc[nt][r];
    }
}

// ------------------------------ utility kernels ----------------------------

__global__ void k_f32_to_f16(const float* __restrict__ in, _Float16* __restrict__ out, int count) {
  for (int i = blockIdx.x * blockDim.x + threadIdx.x; i < count; i += gridDim.x * blockDim.x)
    out[i] = (_Float16)in[i];
}

// dinv[i] = rsqrt(rowsum(A[i,:]) + 2)  (A_hat = A + 2I, A diag is 0)
__global__ __launch_bounds__(256)
void k_rowsum_dinv(const _Float16* __restrict__ A, float* __restrict__ dinv, int n) {
  __shared__ float red[256];
  const int row = blockIdx.x;
  float s = 0.f;
  for (int j = threadIdx.x; j < n; j += 256) s += (float)A[(size_t)row * n + j];
  red[threadIdx.x] = s;
  __syncthreads();
  for (int off = 128; off > 0; off >>= 1) {
    if (threadIdx.x < off) red[threadIdx.x] += red[threadIdx.x + off];
    __syncthreads();
  }
  if (threadIdx.x == 0) {
    const float deg = red[0] + 2.0f;
    dinv[row] = (deg > 0.f) ? rsqrtf(deg) : 0.f;
  }
}

// Z = dinv[:,None] * XW  (f32 + f16 copies)
__global__ void k_make_z(const float* __restrict__ XW, float* __restrict__ Z32,
                         _Float16* __restrict__ Z16, const float* __restrict__ dinv, int count) {
  for (int idx = blockIdx.x * blockDim.x + threadIdx.x; idx < count; idx += gridDim.x * blockDim.x) {
    const float z = dinv[idx >> 6] * XW[idx];
    Z32[idx] = z;
    Z16[idx] = (_Float16)z;
  }
}

// H = dinv[:,None] * (P + 2*Z) + b   [+ relu]
__global__ void k_gcn_finish(const float* __restrict__ P, const float* __restrict__ Z32,
                             const float* __restrict__ dinv, const float* __restrict__ b,
                             float* __restrict__ H, int count, int relu) {
  for (int idx = blockIdx.x * blockDim.x + threadIdx.x; idx < count; idx += gridDim.x * blockDim.x) {
    float v = dinv[idx >> 6] * (P[idx] + 2.0f * Z32[idx]) + b[idx & 63];
    if (relu) v = fmaxf(v, 0.f);
    H[idx] = v;
  }
}

__global__ void k_pnorm(const float* __restrict__ p, float* __restrict__ out) {
  float s = 0.f;
  for (int j = 0; j < 64; ++j) s += p[j] * p[j];
  *out = sqrtf(s);
}

__global__ void k_score(const float* __restrict__ H, const float* __restrict__ p,
                        const float* __restrict__ pnorm, float* __restrict__ score, int n) {
  const int i = blockIdx.x * blockDim.x + threadIdx.x;
  if (i >= n) return;
  float s = 0.f;
  for (int j = 0; j < 64; ++j) s += H[(size_t)i * 64 + j] * p[j];
  score[i] = tanhf(s / (*pnorm));
}

// rank[i] = #{j : s[j] > s[i]  or (s[j]==s[i] and j<i)}; keep if rank < k.
// Matches jax.lax.top_k descending order with lower-index tie-break.
__global__ void k_rank_scatter(const float* __restrict__ score, int* __restrict__ perm, int n, int k) {
  const int i = blockIdx.x * blockDim.x + threadIdx.x;
  if (i >= n) return;
  const float si = score[i];
  int rank = 0;
  for (int j = 0; j < n; ++j) {
    const float sj = score[j];
    rank += (sj > si) || (sj == si && j < i);
  }
  if (rank < k) perm[rank] = i;
}

// x_new[r,:] = H[perm[r],:] * score[perm[r]]
__global__ void k_gather_x(const float* __restrict__ H, const int* __restrict__ perm,
                           const float* __restrict__ score, float* __restrict__ out, int count) {
  for (int idx = blockIdx.x * blockDim.x + threadIdx.x; idx < count; idx += gridDim.x * blockDim.x) {
    const int pi = perm[idx >> 6];
    out[idx] = H[(size_t)pi * 64 + (idx & 63)] * score[pi];
  }
}

__global__ void k_gather_A(const _Float16* __restrict__ Ain, const int* __restrict__ perm,
                           _Float16* __restrict__ Aout, int n, int k) {
  const int c = blockIdx.x * 16 + threadIdx.x;
  const int r = blockIdx.y * 16 + threadIdx.y;
  if (r < k && c < k)
    Aout[(size_t)r * k + c] = Ain[(size_t)perm[r] * n + perm[c]];
}

__global__ void k_copy_f32(const float* __restrict__ a, float* __restrict__ b, int count) {
  for (int i = blockIdx.x * blockDim.x + threadIdx.x; i < count; i += gridDim.x * blockDim.x)
    b[i] = a[i];
}

// Hbig[perm[r],:] += Hsml[r,:]   (perm entries unique -> no atomics needed)
__global__ void k_scatter_add(float* __restrict__ Hbig, const float* __restrict__ Hsml,
                              const int* __restrict__ perm, int count) {
  for (int idx = blockIdx.x * blockDim.x + threadIdx.x; idx < count; idx += gridDim.x * blockDim.x)
    Hbig[(size_t)perm[idx >> 6] * 64 + (idx & 63)] += Hsml[idx];
}

// out[i] = (sum_j relu(H[i,j]) * fc_w[j]) / (1 - prob) + fc_b
__global__ void k_final(const float* __restrict__ H, const float* __restrict__ fc_w,
                        const float* __restrict__ fc_b, const float* __restrict__ prob,
                        float* __restrict__ out, int n) {
  const int i = blockIdx.x * blockDim.x + threadIdx.x;
  if (i >= n) return;
  float s = 0.f;
  for (int j = 0; j < 64; ++j) s += fmaxf(H[(size_t)i * 64 + j], 0.f) * fc_w[j];
  out[i] = s / (1.0f - prob[0]) + fc_b[0];
}

// ---------------------------------------------------------------------------
// Host orchestration.  Workspace requirement: ~85 MB.
// ---------------------------------------------------------------------------
extern "C" void kernel_launch(void* const* d_in, const int* in_sizes, int n_in,
                              void* d_out, int out_size, void* d_ws, size_t ws_size,
                              hipStream_t stream) {
  (void)in_sizes; (void)n_in; (void)out_size; (void)ws_size;

  const float* x      = (const float*)d_in[0];
  const float* adj    = (const float*)d_in[1];
  const float* prob   = (const float*)d_in[2];
  const float* down_W = (const float*)d_in[3];
  const float* down_b = (const float*)d_in[4];
  const float* pool_w = (const float*)d_in[5];
  const float* up_W   = (const float*)d_in[6];
  const float* up_b   = (const float*)d_in[7];
  const float* fc_w   = (const float*)d_in[8];
  const float* fc_b   = (const float*)d_in[9];
  float* out = (float*)d_out;

  const int NL[4] = {4096, 2048, 1024, 512};

  char* ws = (char*)d_ws;
  size_t off = 0;
  auto alloc = [&](size_t bytes) -> void* {
    off = (off + 255) & ~(size_t)255;
    void* p = ws + off;
    off += bytes;
    return p;
  };

  _Float16* Alv[4];
  for (int l = 0; l < 4; ++l) Alv[l] = (_Float16*)alloc((size_t)NL[l] * NL[l] * 2);
  _Float16* A2 = (_Float16*)alloc((size_t)4096 * 4096 * 2);
  float* Xs[3];
  for (int l = 0; l < 3; ++l) Xs[l] = (float*)alloc((size_t)NL[l] * 64 * 4);
  float*     Hbuf  = (float*)alloc((size_t)4096 * 64 * 4);
  float*     Hout  = (float*)alloc((size_t)4096 * 64 * 4);
  float*     C0    = (float*)alloc((size_t)4096 * 64 * 4);
  float*     Z32   = (float*)alloc((size_t)4096 * 64 * 4);
  _Float16*  Z16   = (_Float16*)alloc((size_t)4096 * 64 * 2);
  float*     dinv  = (float*)alloc(4096 * 4);
  float*     score = (float*)alloc(4096 * 4);
  float*     pnorm = (float*)alloc(256);
  int* perms[3];
  for (int l = 0; l < 3; ++l) perms[l] = (int*)alloc((size_t)NL[l + 1] * 4);

  // GCN: H = D^-1/2 (A + 2I) D^-1/2 (X W) + b  [opt relu]
  auto gcn = [&](const float* xin, const _Float16* A, int nloc,
                 const float* W, const float* b, float* hout, int relu) {
    const int cnt = nloc * 64, eb = (cnt + 255) / 256;
    k_rowsum_dinv<<<nloc, 256, 0, stream>>>(A, dinv, nloc);
    k_gemm_n64<true, true><<<nloc / 64, 128, 0, stream>>>((const void*)xin, (const void*)W, C0, nloc, 64, 64);
    k_make_z<<<eb, 256, 0, stream>>>(C0, Z32, Z16, dinv, cnt);
    k_gemm_n64<false, false><<<nloc / 64, 128, 0, stream>>>((const void*)A, (const void*)Z16, C0, nloc, nloc, nloc);
    k_gcn_finish<<<eb, 256, 0, stream>>>(C0, Z32, dinv, b, hout, cnt, relu);
  };

  // ---- down path ----
  k_f32_to_f16<<<4096, 256, 0, stream>>>(adj, Alv[0], 4096 * 4096);
  gcn(x, Alv[0], 4096, down_W, down_b, Xs[0], 1);

  for (int i = 0; i < 3; ++i) {
    const int n = NL[i], k = NL[i + 1];
    dim3 gg(n / 128, n / 128);
    k_augment_gemm<<<gg, 256, 0, stream>>>(Alv[i], A2, n);

    const float* hin = Xs[i];
    k_pnorm<<<1, 1, 0, stream>>>(pool_w + i * 64, pnorm);
    k_score<<<(n + 255) / 256, 256, 0, stream>>>(hin, pool_w + i * 64, pnorm, score, n);
    k_rank_scatter<<<(n + 255) / 256, 256, 0, stream>>>(score, perms[i], n, k);
    k_gather_x<<<(k * 64 + 255) / 256, 256, 0, stream>>>(hin, perms[i], score, Hbuf, k * 64);
    dim3 ga(k / 16, k / 16);
    k_gather_A<<<ga, dim3(16, 16), 0, stream>>>(A2, perms[i], Alv[i + 1], n, k);

    float* tgt = (i < 2) ? Xs[i + 1] : Hout;
    gcn(Hbuf, Alv[i + 1], k, down_W + (size_t)(i + 1) * 64 * 64, down_b + (i + 1) * 64, tgt, 1);
  }

  // ---- up path ----
  for (int i = 0; i < 3; ++i) {
    const int j = 2 - i;
    const int n = NL[j], k = NL[j + 1];
    k_copy_f32<<<(n * 64 + 255) / 256, 256, 0, stream>>>(Xs[j], Hbuf, n * 64);
    k_scatter_add<<<(k * 64 + 255) / 256, 256, 0, stream>>>(Hbuf, Hout, perms[j], k * 64);
    gcn(Hbuf, Alv[j], n, up_W + (size_t)i * 64 * 64, up_b + i * 64, Hout, (i < 2) ? 1 : 0);
  }

  // ---- final relu + dropout(p=prob) + FC ----
  k_final<<<(4096 + 255) / 256, 256, 0, stream>>>(Hout, fc_w, fc_b, prob, out, 4096);
}
